// SelfAttention_28621662061192
// MI455X (gfx1250) — compile-verified, matrix operand order checked by reading, not run on hardware
//
#include <hip/hip_runtime.h>

// ---------------------------------------------------------------------------
// Self-attention B=8, S=2048, E=1024, D=64 on gfx1250 (MI455X).
//   k1: Wq/Wk/Wv fp32 -> f16
//   k2: fused QKV projection (12x v_wmma_f32_16x16x32_f16 per K-step);
//       q,k row-major [B*S,64] f16; v transposed [B,64,S] f16.
//   k3: prep: mask -> fp32 additive bias; V column prefix-sums per 32-block.
//   k4: transposed flash attention:
//         S^T = K*Q^T (lane = i), P^T C-layout == P B-fragment -> no LDS,
//         O^T = V^T*P (lane = i -> scalar softmax state),
//       starting at the diagonal window; the strictly-lower region is applied
//       in closed form (exact for the all-masked uniform-softmax fallback).
// ---------------------------------------------------------------------------

typedef _Float16 v16h __attribute__((ext_vector_type(16)));
typedef _Float16 v8h  __attribute__((ext_vector_type(8)));
typedef float    v8f  __attribute__((ext_vector_type(8)));

#define B_ 8
#define S_ 2048
#define E_ 1024
#define D_ 64

static __device__ __forceinline__ v8f wmma_f16(v16h a, v16h b, v8f c) {
  // D = A(16x32 f16) * B(32x16 f16) + C(16x16 f32)
  return __builtin_amdgcn_wmma_f32_16x16x32_f16(false, a, false, b,
                                                (short)0, c, false, false);
}

static __device__ __forceinline__ v16h load_frag16(const _Float16* p) {
  // 16 halves of one fragment lane: [p+0 .. p+7] and [p+16 .. p+23]
  v16h f;
  ((v8h*)&f)[0] = ((const v8h*)p)[0];
  ((v8h*)&f)[1] = ((const v8h*)p)[2];
  return f;
}

// ---------------------------------------------------------------------------
// Kernel 1: fp32 -> f16 weight conversion (3 x [64,1024], row-major).
// ---------------------------------------------------------------------------
__global__ void __launch_bounds__(256)
attn_cvt_weights(const float* __restrict__ Wq, const float* __restrict__ Wk,
                 const float* __restrict__ Wv, _Float16* __restrict__ wh) {
  int i = blockIdx.x * blockDim.x + threadIdx.x;
  if (i < D_ * E_) {
    wh[i]           = (_Float16)Wq[i];
    wh[D_*E_ + i]   = (_Float16)Wk[i];
    wh[2*D_*E_ + i] = (_Float16)Wv[i];
  }
}

// ---------------------------------------------------------------------------
// Kernel 2: fused QKV projection. One wave per 16-row tile of x, 12
// accumulators (Q/K/V x 4 n-tiles), K-loop over E in steps of 32.
// ---------------------------------------------------------------------------
__global__ void __launch_bounds__(128)
attn_qkv_proj(const float* __restrict__ x, const _Float16* __restrict__ wh,
              const float* __restrict__ bq, const float* __restrict__ bk,
              const float* __restrict__ bv,
              _Float16* __restrict__ qh, _Float16* __restrict__ kh,
              _Float16* __restrict__ vth) {
  const int lane = threadIdx.x & 31;
  const int wave = threadIdx.x >> 5;
  const int tile = blockIdx.x * 4 + wave;   // 0..1023
  const int m0   = tile * 16;
  const int l16  = lane & 15;
  const int half = lane >> 4;
  const int kb   = half * 8;

  v8f acc[12];
  v8f vzero = {};
  #pragma unroll
  for (int i = 0; i < 12; ++i) acc[i] = vzero;

  const float* xrow = x + (size_t)(m0 + l16) * E_;

  for (int k0 = 0; k0 < E_; k0 += 32) {
    v16h a;
    {
      const float4* p0 = (const float4*)(xrow + k0 + kb);
      const float4* p1 = (const float4*)(xrow + k0 + kb + 16);
      float4 f0 = p0[0], f1 = p0[1], f2 = p1[0], f3 = p1[1];
      a[0]=(_Float16)f0.x;  a[1]=(_Float16)f0.y;  a[2]=(_Float16)f0.z;  a[3]=(_Float16)f0.w;
      a[4]=(_Float16)f1.x;  a[5]=(_Float16)f1.y;  a[6]=(_Float16)f1.z;  a[7]=(_Float16)f1.w;
      a[8]=(_Float16)f2.x;  a[9]=(_Float16)f2.y;  a[10]=(_Float16)f2.z; a[11]=(_Float16)f2.w;
      a[12]=(_Float16)f3.x; a[13]=(_Float16)f3.y; a[14]=(_Float16)f3.z; a[15]=(_Float16)f3.w;
    }
    #pragma unroll
    for (int w = 0; w < 3; ++w) {
      const _Float16* wbase = wh + (size_t)w * (D_ * E_);
      #pragma unroll
      for (int n = 0; n < 4; ++n) {
        v16h bf = load_frag16(wbase + (size_t)(n*16 + l16) * E_ + k0 + kb);
        acc[w*4 + n] = wmma_f16(a, bf, acc[w*4 + n]);
      }
    }
  }

  #pragma unroll
  for (int w = 0; w < 3; ++w) {
    const float* bias = (w == 0) ? bq : ((w == 1) ? bk : bv);
    #pragma unroll
    for (int n = 0; n < 4; ++n) {
      const int col = n*16 + l16;
      const float bb = bias[col];
      #pragma unroll
      for (int r = 0; r < 8; ++r) {
        const int rowg = m0 + half*8 + r;
        const float val = acc[w*4 + n][r] + bb;
        if (w == 0) {
          qh[(size_t)rowg * D_ + col] = (_Float16)val;
        } else if (w == 1) {
          kh[(size_t)rowg * D_ + col] = (_Float16)val;
        } else {
          const int bi = rowg >> 11;
          const int s  = rowg & (S_ - 1);
          vth[((size_t)bi * D_ + col) * S_ + s] = (_Float16)val;
        }
      }
    }
  }
}

// ---------------------------------------------------------------------------
// Kernel 3: prep. biasf[b][j] = mask ? 0 : -1e22 (fp32, contiguous per j so
// the flash kernel reads it as broadcast float4s). vcum[b][t][d] = column
// prefix sums of V over j < t*32 (for the closed-form lower-triangle tail).
// ---------------------------------------------------------------------------
__global__ void __launch_bounds__(64)
attn_prep(const int* __restrict__ mask, const _Float16* __restrict__ vth,
          float* __restrict__ biasf, float* __restrict__ vcum) {
  const int b = blockIdx.x;
  const int d = threadIdx.x;          // 0..63
  for (int j = d; j < S_; j += 64)
    biasf[b * S_ + j] = mask[b * S_ + j] ? 0.0f : -1e22f;

  const _Float16* vp = vth + ((size_t)b * D_ + d) * S_;
  float run = 0.0f;
  for (int tblk = 0; tblk < S_ / 32; ++tblk) {
    vcum[((size_t)b * (S_ / 32) + tblk) * D_ + d] = run;
    const v8h* pv = (const v8h*)(vp + tblk * 32);
    #pragma unroll
    for (int g = 0; g < 4; ++g) {
      v8h h = pv[g];
      #pragma unroll
      for (int e = 0; e < 8; ++e) run += (float)h[e];
    }
  }
}

// ---------------------------------------------------------------------------
// Kernel 4: transposed flash attention. grid = (S/16, B), one wave per block.
// ---------------------------------------------------------------------------
template <bool CAUSAL>
static __device__ __forceinline__
void flash_step(int jb, int icol, int rowoff, int kb, int l16,
                const _Float16* __restrict__ kbase,
                const _Float16* __restrict__ vbase,
                const float* __restrict__ bias,
                const v16h& bq0, const v16h& bq1,
                float& mrun, float& lrun, v8f o[4]) {
  const float inv_sqrt_d = 0.125f;   // 1/sqrt(64)

  // S^T = K * Q^T : two 16x16 tiles (j rows), lane = i
  v8f c[2];
  #pragma unroll
  for (int f = 0; f < 2; ++f) {
    const _Float16* kp = kbase + (size_t)(jb + f*16 + l16) * D_;
    v16h ak0 = load_frag16(kp + kb);
    v16h ak1 = load_frag16(kp + 32 + kb);
    v8f cz = {};
    c[f] = wmma_f16(ak0, bq0, cz);
    c[f] = wmma_f16(ak1, bq1, c[f]);
  }

  // mask bias (broadcast, contiguous) + optional causal select; in-lane stats
  float s[2][8];
  float maxv = -INFINITY;
  #pragma unroll
  for (int f = 0; f < 2; ++f) {
    const float4* bp = (const float4*)(bias + jb + f*16 + rowoff);
    float4 bb0 = bp[0], bb1 = bp[1];
    float barr[8] = {bb0.x, bb0.y, bb0.z, bb0.w, bb1.x, bb1.y, bb1.z, bb1.w};
    #pragma unroll
    for (int r = 0; r < 8; ++r) {
      // |score| << 1e22/2^24, so fma(c, inv, -1e22) == -1e22 exactly.
      float sv = fmaf(c[f][r], inv_sqrt_d, barr[r]);
      if (CAUSAL) {
        const int j = jb + f*16 + rowoff + r;
        sv = (j >= icol) ? sv : -1e22f;
      }
      s[f][r] = sv;
      maxv = fmaxf(maxv, sv);
    }
  }
  maxv = fmaxf(maxv, __shfl_xor(maxv, 16, 32));   // other 16 j's, same i

  const float mnew  = fmaxf(mrun, maxv);
  const float scale = __expf(mrun - mnew);

  v16h pfrag;     // P as B-fragment: element e<8 -> f=0, e>=8 -> f=1
  float sum = 0.0f;
  #pragma unroll
  for (int f = 0; f < 2; ++f)
    #pragma unroll
    for (int r = 0; r < 8; ++r) {
      const float e = __expf(s[f][r] - mnew);
      sum += e;
      pfrag[f*8 + r] = (_Float16)e;
    }
  sum += __shfl_xor(sum, 16, 32);

  lrun = lrun * scale + sum;
  mrun = mnew;
  #pragma unroll
  for (int t = 0; t < 4; ++t) o[t] = o[t] * scale;

  // O^T += V^T * P  (A = V^T tile, contiguous; B = P in-register)
  #pragma unroll
  for (int t = 0; t < 4; ++t) {
    v16h av = load_frag16(vbase + (size_t)(t*16 + l16) * S_ + jb + kb);
    o[t] = wmma_f16(av, pfrag, o[t]);
  }
}

__global__ void __launch_bounds__(32)
attn_flash_t(const _Float16* __restrict__ qh, const _Float16* __restrict__ kh,
             const _Float16* __restrict__ vth, const float* __restrict__ biasf,
             const float* __restrict__ vcum, float* __restrict__ out) {
  const int lane   = threadIdx.x & 31;
  const int l16    = lane & 15;
  const int half   = lane >> 4;
  const int kb     = half * 8;
  const int rowoff = half * 8;
  const int i0     = blockIdx.x * 16;
  const int b      = blockIdx.y;
  const int icol   = i0 + l16;          // this lane's query row i
  const int jb0    = i0 & ~31;          // diagonal window start

  // Q^T B-fragments (lane = column i), loaded once, reused every step.
  const _Float16* qp = qh + ((size_t)b * S_ + icol) * D_;
  const v16h bq0 = load_frag16(qp + kb);
  const v16h bq1 = load_frag16(qp + 32 + kb);

  float mrun = -INFINITY;
  float lrun = 0.0f;
  v8f o[4];
  v8f vzero = {};
  #pragma unroll
  for (int t = 0; t < 4; ++t) o[t] = vzero;

  const float* bias = biasf + b * S_;
  const _Float16* kbase = kh + (size_t)b * S_ * D_;
  const _Float16* vbase = vth + (size_t)b * D_ * S_;

  // Diagonal window (only place the causal compare is needed).
  flash_step<true>(jb0, icol, rowoff, kb, l16, kbase, vbase, bias,
                   bq0, bq1, mrun, lrun, o);

  // Remaining windows: j >= i0+16 > every i in tile -> no causal test.
  for (int jb = jb0 + 32; jb < S_; jb += 32) {
    flash_step<false>(jb, icol, rowoff, kb, l16, kbase, vbase, bias,
                      bq0, bq1, mrun, lrun, o);
    if (jb + 32 < S_) {
      __builtin_prefetch(kbase + (size_t)(jb + 32 + l16) * D_, 0, 1);
      __builtin_prefetch(vbase + (size_t)l16 * S_ + jb + 32, 0, 1);
    }
  }

  // Closed-form lower-triangle tail: all j < jb0 are -1e22 for every row.
  // e0 = exp(-1e22 - m): 0 for normal rows, exactly 1 for all-masked rows,
  // reproducing the reference's uniform softmax over all S columns.
  if (jb0 > 0) {
    const float e0 = __expf(-1e22f - mrun);
    lrun += (float)jb0 * e0;
    const float* vc = vcum + ((size_t)b * (S_ / 32) + (jb0 >> 5)) * D_;
    #pragma unroll
    for (int t = 0; t < 4; ++t) {
      const float4* cp = (const float4*)(vc + t*16 + rowoff);
      float4 c0 = cp[0], c1 = cp[1];
      o[t][0] += e0 * c0.x; o[t][1] += e0 * c0.y;
      o[t][2] += e0 * c0.z; o[t][3] += e0 * c0.w;
      o[t][4] += e0 * c1.x; o[t][5] += e0 * c1.y;
      o[t][6] += e0 * c1.z; o[t][7] += e0 * c1.w;
    }
  }

  // Normalize, store: O^T lane = i, vgpr r = d -> contiguous float4 stores.
  const float inv = 1.0f / lrun;   // lrun >= 1 always
  float* op = out + ((size_t)b * S_ + icol) * D_;
  #pragma unroll
  for (int t = 0; t < 4; ++t) {
    float4 w0, w1;
    w0.x = o[t][0]*inv; w0.y = o[t][1]*inv; w0.z = o[t][2]*inv; w0.w = o[t][3]*inv;
    w1.x = o[t][4]*inv; w1.y = o[t][5]*inv; w1.z = o[t][6]*inv; w1.w = o[t][7]*inv;
    float* dst = op + t*16 + rowoff;
    ((float4*)dst)[0] = w0;
    ((float4*)dst)[1] = w1;
  }
}

// ---------------------------------------------------------------------------
extern "C" void kernel_launch(void* const* d_in, const int* in_sizes, int n_in,
                              void* d_out, int out_size, void* d_ws, size_t ws_size,
                              hipStream_t stream) {
  const float* x  = (const float*)d_in[0];
  const int* mask = (const int*)d_in[1];
  const float* Wq = (const float*)d_in[2];
  const float* bq = (const float*)d_in[3];
  const float* Wk = (const float*)d_in[4];
  const float* bk = (const float*)d_in[5];
  const float* Wv = (const float*)d_in[6];
  const float* bv = (const float*)d_in[7];
  float* out = (float*)d_out;

  // Workspace: f16 {weights | q | k | v^T} then f32 {biasf | vcum} ~ 6.9 MB
  _Float16* wh   = (_Float16*)d_ws;
  _Float16* qh   = wh + 3 * D_ * E_;
  _Float16* kh   = qh + (size_t)B_ * S_ * D_;
  _Float16* vth  = kh + (size_t)B_ * S_ * D_;
  float*    biasf = (float*)(vth + (size_t)B_ * S_ * D_);
  float*    vcum  = biasf + (size_t)B_ * S_;

  attn_cvt_weights<<<(D_ * E_ + 255) / 256, 256, 0, stream>>>(Wq, Wk, Wv, wh);
  attn_qkv_proj<<<256, 128, 0, stream>>>(x, wh, bq, bk, bv, qh, kh, vth);
  attn_prep<<<B_, 64, 0, stream>>>(mask, vth, biasf, vcum);
  attn_flash_t<<<dim3(S_ / 16, B_), 32, 0, stream>>>(qh, kh, vth, biasf, vcum, out);
}
